// BottleNeck_17119739641872
// MI455X (gfx1250) — matrix operand order repaired and checked by reading, not compiled, not run on hardware
//
#include <hip/hip_runtime.h>
#include <hip/hip_bf16.h>

typedef __attribute__((ext_vector_type(16))) __bf16 v16bf;
typedef __attribute__((ext_vector_type(8)))  float  v8f;
typedef __attribute__((ext_vector_type(4)))  unsigned int u32x4;
typedef __attribute__((ext_vector_type(8)))  int  i32x8;
typedef __attribute__((ext_vector_type(4)))  int  i32x4;

#define HWSZ   3136
#define IMG_W  56
#define NBATCH 32
#define NPIX   (NBATCH * HWSZ)   // 100352

#if defined(__HIP_DEVICE_COMPILE__) && __has_builtin(__builtin_amdgcn_tensor_load_to_lds)
#define USE_TDM 1
#else
#define USE_TDM 0
#endif

// ----------------------------------------------------------------------------
// Weight prep: per-filter alpha = mean|w|, pack sign(w) as bf16 (+/-1, 0) in
// the contiguous-k A-fragment element order used with v_wmma_f32_16x16x32_bf16:
//   lane L (half h = L>>4, row m = t*16 + (L&15)), element e:
//     Koff = 16h + e (contiguous, same order as the B fragments)
// K ordering for 3x3: K = (ky*3+kx)*CIN + c (matches the conv im2col)
// ----------------------------------------------------------------------------
template<int CIN, int KH>
__global__ void prep_weights(const float* __restrict__ W,
                             __bf16* __restrict__ Apack,
                             float* __restrict__ alpha)
{
    constexpr int E  = CIN * KH * KH;
    constexpr int KT = E / 32;
    const int m   = blockIdx.x;
    const int tid = threadIdx.x;
    const float* w = W + (size_t)m * E;

    __shared__ float red[256];
    float s = 0.f;
    for (int i = tid; i < E; i += 256) s += fabsf(w[i]);
    red[tid] = s;
    __syncthreads();
    for (int off = 128; off > 0; off >>= 1) {
        if (tid < off) red[tid] += red[tid + off];
        __syncthreads();
    }
    if (tid == 0) alpha[m] = red[0] / (float)E;

    const int t = m >> 4, lm = m & 15;
    for (int idx = tid; idx < KT * 32; idx += 256) {
        const int kt = idx >> 5;
        const int h  = (idx >> 4) & 1;
        const int e  = idx & 15;
        const int koff = 16 * h + e;
        const int K = kt * 32 + koff;
        int widx;
        if (KH == 1) {
            widx = K;                       // w[m][c]
        } else {
            const int pos = K / CIN, c = K % CIN;
            const int ky = pos / 3, kx = pos % 3;
            widx = (c * KH + ky) * KH + kx; // w[m][c][ky][kx]
        }
        const float wv = w[widx];
        const float sg = (wv > 0.f) ? 1.f : ((wv < 0.f) ? -1.f : 0.f);
        const int L = h * 16 + lm;
        Apack[(((size_t)t * KT + kt) * 32 + L) * 16 + e] = (__bf16)sg;
    }
}

__global__ void zero_floats(float* __restrict__ p, int n)
{
    const int i = blockIdx.x * 256 + threadIdx.x;
    if (i < n) p[i] = 0.f;
}

// ----------------------------------------------------------------------------
// x: NCHW fp32 -> NHWC bf16 via LDS 32x33 tile transpose (coalesced both ways)
// ----------------------------------------------------------------------------
__global__ void x_to_nhwc_bf16(const float* __restrict__ X, __bf16* __restrict__ XT)
{
    const int n   = blockIdx.x;
    const int hw0 = blockIdx.y * 32;
    const int c0  = blockIdx.z * 32;
    __shared__ float tile[32][33];
    const int tx = threadIdx.x & 31;
    const int ty = threadIdx.x >> 5;

    const float* src = X + ((size_t)n * 256 + c0) * HWSZ + hw0;
#pragma unroll
    for (int r = 0; r < 32; r += 8)
        tile[ty + r][tx] = src[(size_t)(ty + r) * HWSZ + tx];
    __syncthreads();

    __bf16* dst = XT + ((size_t)n * HWSZ + hw0) * 256 + c0;
#pragma unroll
    for (int r = 0; r < 32; r += 8)
        dst[(size_t)(ty + r) * 256 + tx] = (__bf16)tile[tx][ty + r];
}

// ----------------------------------------------------------------------------
// Binary conv as WMMA GEMM on NHWC bf16 activations.
// Block: 256 threads = 8 waves; wave -> 16-pixel N-tile; MT M-tiles per wave.
// Per-block A-pack segment staged to LDS once via the Tensor Data Mover
// (TENSORcnt-tracked DMA), then the K-loop reads A with ds_load_b128 while
// global VMEM streams only the B fragments.
// ----------------------------------------------------------------------------
template<int CIN, int COUT, int KH, int MT, bool NHWC_OUT>
__global__ void bconv_wmma(const __bf16* __restrict__ XT,
                           const __bf16* __restrict__ Apack,
                           float* __restrict__ Y)
{
    constexpr int KT = (CIN * KH * KH) / 32;
    constexpr unsigned AELEMS = (unsigned)MT * KT * 512;   // bf16 elements
    const int wave = threadIdx.x >> 5;
    const int lane = threadIdx.x & 31;
    const int col  = lane & 15;
    const int krow = lane >> 4;
    const int p  = blockIdx.x * 128 + wave * 16 + col;   // global pixel
    const int n  = p / HWSZ;
    const int hw = p % HWSZ;
    const int y  = hw / IMG_W;
    const int x  = hw % IMG_W;
    const int mg = blockIdx.y;

    __shared__ __bf16 Alds[AELEMS];
    const __bf16* Agbl = Apack + (size_t)mg * AELEMS;

#if USE_TDM
    if (wave == 0) {
        // --- Tensor DMA descriptor (D#): 1-D tile, 8-byte elements ---
        const unsigned long long ga = (unsigned long long)(const void*)Agbl;
        const unsigned int ldsoff = (unsigned int)(size_t)(void*)&Alds[0];
        constexpr unsigned int t0 = (AELEMS * 2u) / 8u;  // tile len in 8B units
        u32x4 g0;
        g0[0] = 1u;                                       // count=1, user mode
        g0[1] = ldsoff;                                   // lds_addr
        g0[2] = (unsigned int)(ga & 0xffffffffull);       // global_addr lo
        g0[3] = (unsigned int)((ga >> 32) & 0x01ffffffull) | (2u << 30); // hi + type=2
        i32x8 g1;
        g1[0] = (int)(3u << 16);                          // data_size=3 (8B)
        g1[1] = (int)((t0 & 0xffffu) << 16);              // tensor_dim0[15:0]
        g1[2] = (int)(((t0 >> 16) & 0xffffu) | (1u << 16)); // dim0 hi | dim1=1 lo
        g1[3] = (int)((t0 & 0xffffu) << 16);              // dim1 hi=0 | tile_dim0
        g1[4] = 0;                                        // tile_dim1=0, tile_dim2=0
        g1[5] = (int)t0;                                  // dim0_stride lo32
        g1[6] = (int)((t0 & 0xffffu) << 16);              // stride hi=0 | dim1_stride lo
        g1[7] = (int)((t0 >> 16) & 0xffffu);              // dim1_stride hi
        i32x4 gz;  gz[0] = 0; gz[1] = 0; gz[2] = 0; gz[3] = 0;
#if __clang_major__ >= 23
        i32x8 gz8; gz8[0]=0; gz8[1]=0; gz8[2]=0; gz8[3]=0; gz8[4]=0; gz8[5]=0; gz8[6]=0; gz8[7]=0;
        __builtin_amdgcn_tensor_load_to_lds(g0, g1, gz, gz, gz8, 0);
#else
        __builtin_amdgcn_tensor_load_to_lds(g0, g1, gz, gz, 0);
#endif
        __builtin_amdgcn_s_wait_tensorcnt(0);
    }
    __syncthreads();
#else
    {
        const int4* srcv = (const int4*)Agbl;
        int4* dstv = (int4*)Alds;
        constexpr int NV = (int)(AELEMS * 2u) / 16;
        for (int i = threadIdx.x; i < NV; i += 256) dstv[i] = srcv[i];
        __syncthreads();
    }
#endif

    const __bf16* Al = Alds + (size_t)lane * 16;
    v8f acc[MT] = {};

    for (int kt = 0; kt < KT; ++kt) {
        // ---- B fragment: element e holds K = kt*32 + 16*krow + e (contiguous) ----
        v16bf b;
        if (KH == 1) {
            const int c0 = kt * 32 + krow * 16;
            const __bf16* src = XT + (size_t)p * CIN + c0;
            b = *(const v16bf*)src;
            if (kt + 1 < KT)
                __builtin_prefetch((const void*)(src + 32), 0, 1);
        } else {
            const int pos = kt >> 1;                 // tap index (2 k-tiles/tap)
            const int c0  = (kt & 1) * 32 + krow * 16;
            const int ky  = pos / 3, kx = pos % 3;
            const int yy  = y + ky - 1, xx = x + kx - 1;
            const bool ok = ((unsigned)yy < IMG_W) && ((unsigned)xx < IMG_W);
#pragma unroll
            for (int e = 0; e < 16; ++e) b[e] = (__bf16)0.0f;
            if (ok)
                b = *(const v16bf*)(XT + ((size_t)n * HWSZ + yy * IMG_W + xx) * CIN + c0);
        }
        // ---- MT WMMAs sharing this B fragment; A from LDS ----
#pragma unroll
        for (int t = 0; t < MT; ++t) {
            v16bf a = *(const v16bf*)(Al + ((size_t)t * KT + kt) * 32 * 16);
            acc[t] = __builtin_amdgcn_wmma_f32_16x16x32_bf16(
                false, a, false, b, (short)0, acc[t], false, false);
        }
    }

    if (NHWC_OUT) {
#pragma unroll
        for (int t = 0; t < MT; ++t) {
            const int m0 = mg * (MT * 16) + t * 16 + krow * 8;
            *(v8f*)(Y + (size_t)p * COUT + m0) = acc[t];
        }
    } else {
#pragma unroll
        for (int t = 0; t < MT; ++t) {
#pragma unroll
            for (int r = 0; r < 8; ++r) {
                const int m = mg * (MT * 16) + t * 16 + krow * 8 + r;
                Y[((size_t)n * COUT + m) * HWSZ + hw] = acc[t][r];
            }
        }
    }
}

// ----------------------------------------------------------------------------
// Per-channel sum / sumsq over NHWC raw output (channel = lane -> coalesced).
// ----------------------------------------------------------------------------
template<int C>
__global__ void channel_stats_nhwc(const float* __restrict__ H,
                                   float* __restrict__ sum, float* __restrict__ sq)
{
    constexpr int ROWS = 256 / C;
    const int c = threadIdx.x % C;
    const int r = threadIdx.x / C;
    const int stride = gridDim.x * ROWS;
    float s = 0.f, s2 = 0.f;
    for (int p = blockIdx.x * ROWS + r; p < NPIX; p += stride) {
        const float v = H[(size_t)p * C + c];
        s += v; s2 += v * v;
    }
    __shared__ float rs[256], rq[256];
    rs[threadIdx.x] = s; rq[threadIdx.x] = s2;
    __syncthreads();
    for (int g = ROWS >> 1; g > 0; g >>= 1) {
        if (r < g) {
            rs[threadIdx.x] += rs[threadIdx.x + g * C];
            rq[threadIdx.x] += rq[threadIdx.x + g * C];
        }
        __syncthreads();
    }
    if (r == 0) {
        atomicAdd(&sum[c], rs[c]);
        atomicAdd(&sq[c], rq[c]);
    }
}

// Per-channel stats over NCHW raw (stage 3): one block per (n,c) slice.
__global__ void channel_stats_nchw(const float* __restrict__ H, int C,
                                   float* __restrict__ sum, float* __restrict__ sq)
{
    const int b = blockIdx.x;        // n*C + c
    const int c = b % C;
    const float* base = H + (size_t)b * HWSZ;
    float s = 0.f, s2 = 0.f;
    for (int i = threadIdx.x; i < HWSZ; i += 256) {
        const float v = base[i];
        s += v; s2 += v * v;
    }
    __shared__ float rs[256], rq[256];
    rs[threadIdx.x] = s; rq[threadIdx.x] = s2;
    __syncthreads();
    for (int off = 128; off > 0; off >>= 1) {
        if (threadIdx.x < off) {
            rs[threadIdx.x] += rs[threadIdx.x + off];
            rq[threadIdx.x] += rq[threadIdx.x + off];
        }
        __syncthreads();
    }
    if (threadIdx.x == 0) {
        atomicAdd(&sum[c], rs[0]);
        atomicAdd(&sq[c], rq[0]);
    }
}

// Fold alpha + batch stats + gamma/beta into per-channel scale/shift.
__global__ void finalize_stats(const float* __restrict__ sum, const float* __restrict__ sq,
                               const float* __restrict__ alpha,
                               const float* __restrict__ gamma, const float* __restrict__ beta,
                               int C, float* __restrict__ scale, float* __restrict__ shift)
{
    const int c = threadIdx.x;
    if (c < C) {
        const float inv_cnt = 1.f / (float)NPIX;
        const float a  = alpha[c];
        const float mr = sum[c] * inv_cnt;
        const float vr = sq[c] * inv_cnt - mr * mr;
        const float mu = a * mr, var = a * a * vr;
        const float rr = rsqrtf(var + 1e-5f);
        scale[c] = a * rr * gamma[c];
        shift[c] = beta[c] - mu * rr * gamma[c];
    }
}

// BN + hardtanh on NHWC raw -> NHWC bf16 activations
template<int C>
__global__ void bn_act_nhwc(const float* __restrict__ H,
                            const float* __restrict__ scale, const float* __restrict__ shift,
                            __bf16* __restrict__ out)
{
    const size_t i = (size_t)blockIdx.x * 256 + threadIdx.x;
    const int c = (int)(i & (C - 1));
    float yv = H[i] * scale[c] + shift[c];
    yv = fminf(fmaxf(yv, -1.f), 1.f);
    out[i] = (__bf16)yv;
}

// Final: BN + residual + hardtanh, in place on d_out (NCHW raw conv3)
__global__ void bn_final(float* __restrict__ HrawOut, const float* __restrict__ X,
                         const float* __restrict__ scale, const float* __restrict__ shift)
{
    const size_t i = (size_t)blockIdx.x * 256 + threadIdx.x;
    const int c = (int)((i / HWSZ) % 256);
    float yv = HrawOut[i] * scale[c] + shift[c] + X[i];
    HrawOut[i] = fminf(fmaxf(yv, -1.f), 1.f);
}

// ----------------------------------------------------------------------------
extern "C" void kernel_launch(void* const* d_in, const int* in_sizes, int n_in,
                              void* d_out, int out_size, void* d_ws, size_t ws_size,
                              hipStream_t stream)
{
    (void)in_sizes; (void)n_in; (void)out_size; (void)ws_size;

    const float* x  = (const float*)d_in[0];
    const float* w1 = (const float*)d_in[1];
    const float* g1 = (const float*)d_in[2];
    const float* b1 = (const float*)d_in[3];
    const float* w2 = (const float*)d_in[4];
    const float* g2 = (const float*)d_in[5];
    const float* b2 = (const float*)d_in[6];
    const float* w3 = (const float*)d_in[7];
    const float* g3 = (const float*)d_in[8];
    const float* b3 = (const float*)d_in[9];
    float* out = (float*)d_out;

    char* ws = (char*)d_ws;
    size_t off = 0;
    auto take = [&](size_t bytes) -> void* {
        void* p = ws + off;
        off = (off + bytes + 255) & ~(size_t)255;
        return p;
    };
    __bf16* Apack1 = (__bf16*)take(4  * 8  * 32 * 16 * 2);
    __bf16* Apack2 = (__bf16*)take(4  * 18 * 32 * 16 * 2);
    __bf16* Apack3 = (__bf16*)take(16 * 2  * 32 * 16 * 2);
    float*  alpha1 = (float*)take(64 * 4);
    float*  alpha2 = (float*)take(64 * 4);
    float*  alpha3 = (float*)take(256 * 4);
    float*  stats  = (float*)take(768 * 4);
    float* sum1 = stats,       *sq1 = stats + 64;
    float* sum2 = stats + 128, *sq2 = stats + 192;
    float* sum3 = stats + 256, *sq3 = stats + 512;
    float*  coef   = (float*)take(768 * 4);
    float* sc1 = coef,       *sh1 = coef + 64;
    float* sc2 = coef + 128, *sh2 = coef + 192;
    float* sc3 = coef + 256, *sh3 = coef + 512;

    __bf16* xT   = (__bf16*)take((size_t)NPIX * 256 * 2);        // ~51.4 MB
    const size_t mid_elems = (size_t)NPIX * 64;
    float*  h_raw = (float*)take(mid_elems * 4);                 // ~25.7 MB
    __bf16* act1  = (__bf16*)take(mid_elems * 2);                // ~12.9 MB
    __bf16* act2  = (__bf16*)take(mid_elems * 2);                // ~12.9 MB

    // ---- prep ----
    prep_weights<256, 1><<<64,  256, 0, stream>>>(w1, Apack1, alpha1);
    prep_weights<64,  3><<<64,  256, 0, stream>>>(w2, Apack2, alpha2);
    prep_weights<64,  1><<<256, 256, 0, stream>>>(w3, Apack3, alpha3);
    zero_floats<<<3, 256, 0, stream>>>(stats, 768);
    x_to_nhwc_bf16<<<dim3(NBATCH, 98, 8), 256, 0, stream>>>(x, xT);

    // ---- stage 1: 1x1 conv 256->64 (MT=4, LDS A = 32 KB) ----
    bconv_wmma<256, 64, 1, 4, true><<<dim3(784, 1), 256, 0, stream>>>(xT, Apack1, h_raw);
    channel_stats_nhwc<64><<<1024, 256, 0, stream>>>(h_raw, sum1, sq1);
    finalize_stats<<<1, 256, 0, stream>>>(sum1, sq1, alpha1, g1, b1, 64, sc1, sh1);
    bn_act_nhwc<64><<<25088, 256, 0, stream>>>(h_raw, sc1, sh1, act1);

    // ---- stage 2: 3x3 conv 64->64 (MT=2 x grid.y=2, LDS A = 36 KB) ----
    bconv_wmma<64, 64, 3, 2, true><<<dim3(784, 2), 256, 0, stream>>>(act1, Apack2, h_raw);
    channel_stats_nhwc<64><<<1024, 256, 0, stream>>>(h_raw, sum2, sq2);
    finalize_stats<<<1, 256, 0, stream>>>(sum2, sq2, alpha2, g2, b2, 64, sc2, sh2);
    bn_act_nhwc<64><<<25088, 256, 0, stream>>>(h_raw, sc2, sh2, act2);

    // ---- stage 3: 1x1 conv 64->256 (MT=4 x grid.y=4, LDS A = 8 KB) ----
    bconv_wmma<64, 256, 1, 4, false><<<dim3(784, 4), 256, 0, stream>>>(act2, Apack3, out);
    channel_stats_nchw<<<NBATCH * 256, 256, 0, stream>>>(out, 256, sum3, sq3);
    finalize_stats<<<1, 256, 0, stream>>>(sum3, sq3, alpha3, g3, b3, 256, sc3, sh3);
    bn_final<<<100352, 256, 0, stream>>>(out, x, sc3, sh3);
}
